// LSTMRiskRegressor_16492674417244
// MI455X (gfx1250) — matrix operand order repaired and checked by reading, not compiled, not run on hardware
//
#include <hip/hip_runtime.h>
#include <cstddef>

// ---- types for CDNA5 WMMA ----
typedef _Float16 h16;
typedef __attribute__((ext_vector_type(16))) _Float16 v16h;
typedef __attribute__((ext_vector_type(8)))  _Float16 v8h;
typedef __attribute__((ext_vector_type(8)))  float    v8f;

#define B_  512
#define T_  1024
#define F_  32
#define H_  128
#define G4  512   // 4*H

// ws layout (f16 elements)
#define OFF_WIH0 0          // 512x32
#define OFF_WHH0 16384      // 512x128
#define OFF_WIH1 81920      // 512x128
#define OFF_WHH1 147456     // 512x128
#define WTOTAL   212992

// dynamic LDS layout (bytes)
#define L_GATES  0          // 16*512*4 = 32768
#define L_H0     32768      // 16*128*2 = 4096
#define L_H1     36864      // 4096
#define L_XS     40960      // 16*32*2  = 1024
#define L_B0     41984      // 512*4    = 2048
#define L_B1     44032      // 2048
#define L_WIH0   46080      // 512*32*2 = 32768
#define SMEM_BYTES 78848

__device__ __forceinline__ float sigf(float x) { return 1.0f / (1.0f + __expf(-x)); }

// A-fragment (16x32 f16, M x K) from row-major [16][ldk] f16 buffer at K offset kb.
// ISA 7.12.2: lanes 0-15 (M=lane): v0..3 = K 0..7, v4..7 = K 16..23 ;
//             lanes 16-31: v0..3 = K 8..15, v4..7 = K 24..31.
__device__ __forceinline__ v16h load_afrag(const h16* base, int ldk, int kb, int lane) {
  const int m  = lane & 15;
  const int hi = lane >> 4;
  const h16* p0 = base + m * ldk + kb + hi * 8;
  const h16* p1 = base + m * ldk + kb + 16 + hi * 8;
  v8h lo = *(const v8h*)p0;
  v8h hh = *(const v8h*)p1;
  v16h a;
#pragma unroll
  for (int i = 0; i < 8; ++i) { a[i] = lo[i]; a[8 + i] = hh[i]; }
  return a;
}

// B-fragment (32x16 f16, K x N) where B[k][n] = W[n][k], W row-major [N][ldk].
// lanes 0-15: column n0+lane, K = kb+0..7,16..23 ; lanes 16-31: column n0+lane-16, K = kb+8..15,24..31.
__device__ __forceinline__ v16h load_bfrag(const h16* w, int ldk, int n0, int kb, int lane) {
  const int n  = n0 + (lane & 15);
  const int hi = lane >> 4;
  const h16* p = w + n * ldk + kb + hi * 8;
  v8h a = *(const v8h*)p;
  v8h b = *(const v8h*)(p + 16);
  v16h r;
#pragma unroll
  for (int i = 0; i < 8; ++i) { r[i] = a[i]; r[8 + i] = b[i]; }
  return r;
}

__device__ __forceinline__ v8f wmma_f16(v16h a, v16h b, v8f c) {
  return __builtin_amdgcn_wmma_f32_16x16x32_f16(false, a, false, b, (short)0, c, false, false);
}

// ---------------- prep: fp32 weights -> f16 in workspace ----------------
__global__ __launch_bounds__(256) void convert_weights(
    const float* __restrict__ wih0, const float* __restrict__ whh0,
    const float* __restrict__ wih1, const float* __restrict__ whh1,
    h16* __restrict__ ws) {
  int i = blockIdx.x * 256 + threadIdx.x;
  if (i < OFF_WHH0)       ws[i] = (h16)wih0[i - OFF_WIH0];
  else if (i < OFF_WIH1)  ws[i] = (h16)whh0[i - OFF_WHH0];
  else if (i < OFF_WHH1)  ws[i] = (h16)wih1[i - OFF_WIH1];
  else if (i < WTOTAL)    ws[i] = (h16)whh1[i - OFF_WHH1];
}

// ---------------- fused 2-layer LSTM + head, one WG per 16 batch rows ----------------
// Per-wave weight B-fragments for Whh0 / Wih1 / Whh1 are loop-invariant and live in
// registers (48 x v16h = 384 VGPRs); only A-fragments stream through LDS each step.
__global__ __launch_bounds__(256) void lstm_fused(
    const float* __restrict__ x,            // (B,T,F) fp32
    const h16*  __restrict__ wih0g,         // 512x32 f16
    const h16*  __restrict__ whh0g,         // 512x128 f16
    const h16*  __restrict__ wih1g,         // 512x128 f16
    const h16*  __restrict__ whh1g,         // 512x128 f16
    const float* __restrict__ bih0, const float* __restrict__ bhh0,
    const float* __restrict__ bih1, const float* __restrict__ bhh1,
    const float* __restrict__ ln_g, const float* __restrict__ ln_b,
    const float* __restrict__ W1,   const float* __restrict__ b1,
    const float* __restrict__ W2,   const float* __restrict__ b2,
    float* __restrict__ out) {
  extern __shared__ uint4 smem4[];
  char* smem = (char*)smem4;
  float* gates = (float*)(smem + L_GATES);   // [16][512]
  h16*   h0s   = (h16*)(smem + L_H0);        // [16][128]
  h16*   h1s   = (h16*)(smem + L_H1);        // [16][128]
  h16*   xs    = (h16*)(smem + L_XS);        // [16][32]
  float* b0s   = (float*)(smem + L_B0);      // [512]
  float* b1s   = (float*)(smem + L_B1);      // [512]
  h16*   wih0s = (h16*)(smem + L_WIH0);      // [512][32]

  const int tid  = threadIdx.x;
  const int lane = tid & 31;
  const int wv   = tid >> 5;                 // wave 0..7 -> gate columns wv*64..+63
  const int brow = blockIdx.x * 16;
  const int n0w  = wv * 64;

  // stage Wih0 into LDS, fuse biases, zero h
  {
    const uint4* s0 = (const uint4*)wih0g; uint4* d0 = (uint4*)wih0s;
#pragma unroll 2
    for (int i = tid; i < 32768 / 16; i += 256) d0[i] = s0[i];
  }
  for (int i = tid; i < G4; i += 256) {
    b0s[i] = bih0[i] + bhh0[i];
    b1s[i] = bih1[i] + bhh1[i];
  }
  for (int i = tid; i < 16 * H_; i += 256) { h0s[i] = (h16)0.0f; h1s[i] = (h16)0.0f; }

  // ---- loop-invariant weight B-fragments -> registers (384 VGPRs) ----
  v16h wHH0[16], wIH1[16], wHH1[16];
#pragma unroll
  for (int kb = 0; kb < 4; ++kb) {
#pragma unroll
    for (int nt = 0; nt < 4; ++nt) {
      wHH0[kb * 4 + nt] = load_bfrag(whh0g, H_, n0w + nt * 16, kb * 32, lane);
      wIH1[kb * 4 + nt] = load_bfrag(wih1g, H_, n0w + nt * 16, kb * 32, lane);
      wHH1[kb * 4 + nt] = load_bfrag(whh1g, H_, n0w + nt * 16, kb * 32, lane);
    }
  }
  __syncthreads();

  // persistent cell state: thread -> (row em, 8 columns starting at ej)
  const int em = tid >> 4;
  const int ej = (tid & 15) * 8;
  float c0r[8], c1r[8];
#pragma unroll
  for (int i = 0; i < 8; ++i) { c0r[i] = 0.0f; c1r[i] = 0.0f; }

  const int mh = (lane >> 4) * 8;

  for (int t = 0; t < T_; ++t) {
    // ---- stage x tile (16 rows x 32 cols) as f16; prefetch next step ----
    for (int i = tid; i < 16 * F_; i += 256) {
      int m = i >> 5, f = i & 31;
      const float* px = &x[((size_t)(brow + m) * T_ + t) * F_ + f];
      xs[i] = (h16)(*px);
      if (t + 1 < T_) __builtin_prefetch(px + F_, 0, 1);
    }
    __syncthreads();

    // ---- layer 0 gates: gates = x_t @ Wih0^T + h0 @ Whh0^T ----
    {
      v8f acc[4];
      v16h ax = load_afrag(xs, F_, 0, lane);
#pragma unroll
      for (int nt = 0; nt < 4; ++nt) {
        v16h b = load_bfrag(wih0s, F_, n0w + nt * 16, 0, lane);
        acc[nt] = wmma_f16(ax, b, (v8f){0, 0, 0, 0, 0, 0, 0, 0});
      }
#pragma unroll
      for (int kb = 0; kb < 4; ++kb) {
        v16h ah = load_afrag(h0s, H_, kb * 32, lane);
#pragma unroll
        for (int nt = 0; nt < 4; ++nt)
          acc[nt] = wmma_f16(ah, wHH0[kb * 4 + nt], acc[nt]);
      }
#pragma unroll
      for (int nt = 0; nt < 4; ++nt) {
        int n = n0w + nt * 16 + (lane & 15);
#pragma unroll
        for (int r = 0; r < 8; ++r) gates[(mh + r) * G4 + n] = acc[nt][r];
      }
    }
    __syncthreads();

    // ---- layer 0 cell update ----
#pragma unroll
    for (int q = 0; q < 8; ++q) {
      int j = ej + q;
      float ig = gates[em * G4 + j]           + b0s[j];
      float fg = gates[em * G4 + H_ + j]      + b0s[H_ + j];
      float gg = gates[em * G4 + 2 * H_ + j]  + b0s[2 * H_ + j];
      float og = gates[em * G4 + 3 * H_ + j]  + b0s[3 * H_ + j];
      float c  = sigf(fg) * c0r[q] + sigf(ig) * tanhf(gg);
      c0r[q]   = c;
      h0s[em * H_ + j] = (h16)(sigf(og) * tanhf(c));
    }
    __syncthreads();

    // ---- layer 1 gates: gates = h0 @ Wih1^T + h1 @ Whh1^T ----
    {
      v8f acc[4];
#pragma unroll
      for (int nt = 0; nt < 4; ++nt) acc[nt] = (v8f){0, 0, 0, 0, 0, 0, 0, 0};
#pragma unroll
      for (int kb = 0; kb < 4; ++kb) {
        v16h a0 = load_afrag(h0s, H_, kb * 32, lane);
        v16h a1 = load_afrag(h1s, H_, kb * 32, lane);
#pragma unroll
        for (int nt = 0; nt < 4; ++nt) {
          acc[nt] = wmma_f16(a0, wIH1[kb * 4 + nt], acc[nt]);
          acc[nt] = wmma_f16(a1, wHH1[kb * 4 + nt], acc[nt]);
        }
      }
#pragma unroll
      for (int nt = 0; nt < 4; ++nt) {
        int n = n0w + nt * 16 + (lane & 15);
#pragma unroll
        for (int r = 0; r < 8; ++r) gates[(mh + r) * G4 + n] = acc[nt][r];
      }
    }
    __syncthreads();

    // ---- layer 1 cell update (no trailing barrier needed: xs is not re-read
    //      until after the next iteration's barrier, and h1s/gates consumers
    //      are separated by >=1 barrier) ----
#pragma unroll
    for (int q = 0; q < 8; ++q) {
      int j = ej + q;
      float ig = gates[em * G4 + j]           + b1s[j];
      float fg = gates[em * G4 + H_ + j]      + b1s[H_ + j];
      float gg = gates[em * G4 + 2 * H_ + j]  + b1s[2 * H_ + j];
      float og = gates[em * G4 + 3 * H_ + j]  + b1s[3 * H_ + j];
      float c  = sigf(fg) * c1r[q] + sigf(ig) * tanhf(gg);
      c1r[q]   = c;
      h1s[em * H_ + j] = (h16)(sigf(og) * tanhf(c));
    }
  }
  __syncthreads();

  // ---- head: LayerNorm(H) -> Linear(H,64) -> ReLU -> Linear(64,1) -> sigmoid ----
  float* lnv = gates;          // reuse: 16*128 f32
  float* hid = gates + 2048;   // 16*64 f32
  if (tid < 16) {
    int m = tid;
    float s = 0.0f;
    for (int j = 0; j < H_; ++j) s += (float)h1s[m * H_ + j];
    float mu = s * (1.0f / H_);
    float v = 0.0f;
    for (int j = 0; j < H_; ++j) { float d = (float)h1s[m * H_ + j] - mu; v += d * d; }
    float rs = rsqrtf(v * (1.0f / H_) + 1e-5f);
    for (int j = 0; j < H_; ++j)
      lnv[m * H_ + j] = ((float)h1s[m * H_ + j] - mu) * rs * ln_g[j] + ln_b[j];
  }
  __syncthreads();
  for (int idx = tid; idx < 16 * 64; idx += 256) {
    int m = idx >> 6, o = idx & 63;
    float s = b1[o];
    for (int j = 0; j < H_; ++j) s += lnv[m * H_ + j] * W1[o * H_ + j];
    hid[m * 64 + o] = s > 0.0f ? s : 0.0f;
  }
  __syncthreads();
  if (tid < 16) {
    int m = tid;
    float s = b2[0];
    for (int o = 0; o < 64; ++o) s += hid[m * 64 + o] * W2[o];
    out[brow + m] = 1.0f / (1.0f + __expf(-s));
  }
}

extern "C" void kernel_launch(void* const* d_in, const int* in_sizes, int n_in,
                              void* d_out, int out_size, void* d_ws, size_t ws_size,
                              hipStream_t stream) {
  (void)in_sizes; (void)n_in; (void)out_size; (void)ws_size;
  const float* x    = (const float*)d_in[0];
  const float* Wih0 = (const float*)d_in[1];
  const float* Whh0 = (const float*)d_in[2];
  const float* bih0 = (const float*)d_in[3];
  const float* bhh0 = (const float*)d_in[4];
  const float* Wih1 = (const float*)d_in[5];
  const float* Whh1 = (const float*)d_in[6];
  const float* bih1 = (const float*)d_in[7];
  const float* bhh1 = (const float*)d_in[8];
  const float* ln_g = (const float*)d_in[9];
  const float* ln_b = (const float*)d_in[10];
  const float* W1   = (const float*)d_in[11];
  const float* b1   = (const float*)d_in[12];
  const float* W2   = (const float*)d_in[13];
  const float* b2   = (const float*)d_in[14];

  h16* wf = (h16*)d_ws;
  convert_weights<<<(WTOTAL + 255) / 256, 256, 0, stream>>>(Wih0, Whh0, Wih1, Whh1, wf);
  lstm_fused<<<B_ / 16, 256, SMEM_BYTES, stream>>>(
      x, wf + OFF_WIH0, wf + OFF_WHH0, wf + OFF_WIH1, wf + OFF_WHH1,
      bih0, bhh0, bih1, bhh1, ln_g, ln_b, W1, b1, W2, b2, (float*)d_out);
}
